// Stage3_56212531970264
// MI455X (gfx1250) — matrix operand order repaired and checked by reading, not compiled, not run on hardware
//
#include <hip/hip_runtime.h>
#include <hip/hip_bf16.h>
#include <stdint.h>

// Problem dims
#define T_SEQ 128
#define B_SZ  128
#define H_SZ  1024
#define I_SZ  2048
#define G4H   4096
#define ROWS  (T_SEQ * B_SZ)   // 16384

// h(t) global staging: parity p base = p * STG_PAR; hi at +0, lo at +STG_HALF
#define STG_HALF (B_SZ * H_SZ)        // 131072 bf16
#define STG_PAR  (2 * STG_HALF)       // 262144 bf16 per parity
#define STG_TOT  (2 * STG_PAR)        // 524288 bf16 = 1 MB per layer

typedef __bf16 bf16;
typedef __bf16 v16bf __attribute__((ext_vector_type(16)));
typedef __bf16 v8bf  __attribute__((ext_vector_type(8)));
typedef float  v8f   __attribute__((ext_vector_type(8)));

static __device__ __forceinline__ v8f wmma_bf16(v16bf a, v16bf b, v8f c) {
  // D = A x B + C, f32 accumulate, 16x16x32 bf16 (wave32)
  return __builtin_amdgcn_wmma_f32_16x16x32_bf16(false, a, false, b, (short)0, c,
                                                 false, false);
}

static __device__ __forceinline__ float sigmoidf_(float x) {
  return 1.0f / (1.0f + __expf(-x));
}

// Async copy 16 bytes global -> LDS (ASYNCcnt-tracked, no VGPR round trip).
static __device__ __forceinline__ void async_cp16(uint32_t lds_off,
                                                  const void* gsrc) {
  asm volatile("global_load_async_to_lds_b128 %0, %1, off"
               :: "v"(lds_off), "v"((unsigned long long)(uintptr_t)gsrc)
               : "memory");
}
static __device__ __forceinline__ void async_wait0() {
  asm volatile("s_wait_asynccnt 0" ::: "memory");
}

// ---------------------------------------------------------------------------
// Zero a u32 region (staging parities + slab sync counters, each call).
// ---------------------------------------------------------------------------
__global__ __launch_bounds__(256) void zero32_kernel(uint32_t* __restrict__ p,
                                                     int n) {
  for (int i = blockIdx.x * blockDim.x + threadIdx.x; i < n;
       i += gridDim.x * blockDim.x)
    p[i] = 0u;
}

// ---------------------------------------------------------------------------
// Split f32 weights into bf16 hi/lo pair (error-compensated bf16x2 GEMM).
// ---------------------------------------------------------------------------
__global__ __launch_bounds__(256) void cvt_split_kernel(
    const float* __restrict__ w, bf16* __restrict__ hi, bf16* __restrict__ lo,
    int n) {
  for (int i = blockIdx.x * blockDim.x + threadIdx.x; i < n;
       i += gridDim.x * blockDim.x) {
    float x = w[i];
    bf16 h = (bf16)x;
    hi[i] = h;
    lo[i] = (bf16)(x - (float)h);
  }
}

// ---------------------------------------------------------------------------
// out1 = input4 ; out14[:, :, H:2H] = input4  (concat halves that are static)
// ---------------------------------------------------------------------------
__global__ __launch_bounds__(256) void prep_out_kernel(
    const float* __restrict__ in4, float* __restrict__ out1,
    float* __restrict__ out14) {
  const size_t n = (size_t)ROWS * H_SZ;
  for (size_t i = blockIdx.x * (size_t)blockDim.x + threadIdx.x; i < n;
       i += (size_t)gridDim.x * blockDim.x) {
    float v = in4[i];
    out1[i] = v;
    size_t r = i >> 10;          // row = t*B + b
    size_t j = i & 1023;
    out14[r * 2048 + 1024 + j] = v;
  }
}

// ---------------------------------------------------------------------------
// Input projection GEMM:
//   XP[ROWS,4096] = concat(X0, X1)[ROWS,2048] @ W^T + (b_ih + b_hh)
// A is f32 in memory, split to bf16 hi/lo on the fly; W pre-split in ws.
// Workgroup tile 64(M) x 256(N); 8 waves, each 16x128 (8 subtiles).
// ---------------------------------------------------------------------------
__global__ __launch_bounds__(256) void gemm_xproj_kernel(
    const float* __restrict__ X0,   // [ROWS, 1024]  (first concat half)
    const float* __restrict__ X1,   // [ROWS, 1024]  (second concat half)
    const bf16* __restrict__ Whi,   // [4096, 2048] row-major bf16 hi
    const bf16* __restrict__ Wlo,   // [4096, 2048] row-major bf16 lo
    const float* __restrict__ bih, const float* __restrict__ bhh,
    float* __restrict__ XP)         // [ROWS, 4096]
{
  const int lane = threadIdx.x & 31;
  const int wave = threadIdx.x >> 5;
  const int hl = lane >> 4;        // half-lane selector
  const int ln = lane & 15;
  const int wm = wave >> 1;        // 0..3
  const int wn = wave & 1;         // 0..1
  const int mBase = blockIdx.y * 64 + wm * 16;
  const int nBase = blockIdx.x * 256 + wn * 128;
  const int row = mBase + ln;      // A-row owned by this lane (both halves)

  v8f acc[8] = {};

  for (int k0 = 0; k0 < I_SZ; k0 += 32) {
    // --- A tile: 16 f32 per lane at K = k0 + hl*16 .. +16
    const float* src = (k0 < H_SZ)
                           ? (X0 + (size_t)row * H_SZ + k0)
                           : (X1 + (size_t)row * H_SZ + (k0 - H_SZ));
    float4 f0 = *(const float4*)(src + hl * 16);
    float4 f1 = *(const float4*)(src + hl * 16 + 4);
    float4 f2 = *(const float4*)(src + hl * 16 + 8);
    float4 f3 = *(const float4*)(src + hl * 16 + 12);
    float xa[16] = {f0.x, f0.y, f0.z, f0.w, f1.x, f1.y, f1.z, f1.w,
                    f2.x, f2.y, f2.z, f2.w, f3.x, f3.y, f3.z, f3.w};
    v16bf ahi, alo;
#pragma unroll
    for (int i = 0; i < 16; ++i) {
      bf16 h = (bf16)xa[i];
      ahi[i] = h;
      alo[i] = (bf16)(xa[i] - (float)h);
    }

#pragma unroll
    for (int s = 0; s < 8; ++s) {
      const int g = nBase + s * 16 + ln;   // output column == weight row
      const v16bf Bh = *(const v16bf*)(Whi + (size_t)g * I_SZ + k0 + hl * 16);
      const v16bf Bl = *(const v16bf*)(Wlo + (size_t)g * I_SZ + k0 + hl * 16);
      acc[s] = wmma_bf16(ahi, Bh, acc[s]);
      acc[s] = wmma_bf16(ahi, Bl, acc[s]);
      acc[s] = wmma_bf16(alo, Bh, acc[s]);
    }
  }

#pragma unroll
  for (int s = 0; s < 8; ++s) {
    const int g = nBase + s * 16 + ln;
    const float bias = bih[g] + bhh[g];
#pragma unroll
    for (int r = 0; r < 8; ++r) {
      XP[(size_t)(mBase + r + hl * 8) * G4H + g] = acc[s][r] + bias;
    }
  }
}

// ---------------------------------------------------------------------------
// LSTM recurrence, slab-parallel. 64 workgroups:
//   wgB = blockIdx.x>>3 : batch slab (16 rows)   — independent recurrences
//   wgJ = blockIdx.x&7  : hidden slice (128 of 1024 units)
// Each WG (256 thr, 8 waves) computes the 4 gate tiles (g = gate*1024+j) of
// its 16-wide j-subtiles so i,f,g,o land in its own accumulators and the cell
// update is register-local (c stays in VGPRs across all T steps).
// h(t) is exchanged between the 8 WGs of a slab through L2 (bf16 hi/lo,
// ping-pong) with one release/acquire counter barrier per timestep. Each WG
// pulls its slab's h(t-1) into LDS via global_load_async_to_lds_b128.
// ---------------------------------------------------------------------------
__global__ __launch_bounds__(256) void lstm_rec_kernel(
    const float* __restrict__ XP,   // [ROWS, 4096] pre-activations (x-proj)
    const bf16* __restrict__ Uhi,   // [4096, 1024] w_hh bf16 hi
    const bf16* __restrict__ Ulo,   // [4096, 1024] w_hh bf16 lo
    const float* __restrict__ Hres, // layer2: h1 (residual); layer1: unused
    float* __restrict__ Hout,       // layer1: h1 ; layer2: out11 region
    float* __restrict__ Out14,      // layer2: out14 region ; layer1: unused
    bf16* __restrict__ stage,       // [2][{hi,lo}][128][1024] (parity0 zeroed)
    uint32_t* __restrict__ sync,    // 8 counters, one per slab (pre-zeroed)
    int layer2) {
  __shared__ bf16 sA[2 * 16 * H_SZ];   // [hi 16x1024 | lo 16x1024] = 64 KB

  const int lane = threadIdx.x & 31;
  const int wave = threadIdx.x >> 5;   // 0..7
  const int hl = lane >> 4;
  const int ln = lane & 15;
  const int wgB = blockIdx.x >> 3;     // batch slab
  const int wgJ = blockIdx.x & 7;      // hidden slice
  const int bBase = wgB * 16;
  const int j = wgJ * 128 + wave * 16 + ln;  // hidden unit for this lane
  uint32_t* ctr = sync + wgB;
  const uint32_t ldsBase = (uint32_t)(uintptr_t)&sA[0];

  v8f cst = {};  // cell state for this wave's 16b x 16j tile

  for (int t = 0; t < T_SEQ; ++t) {
    const int rp = t & 1;          // read parity (t=0 -> zeroed parity 0)
    const int wp = (t + 1) & 1;    // write parity

    // ---- stage h(t-1) hi/lo of our slab: global(L2) -> LDS, async
    __syncthreads();  // previous step's LDS readers are done
    {
      const bf16* srcHi = stage + (size_t)rp * STG_PAR + bBase * H_SZ;
      const bf16* srcLo = srcHi + STG_HALF;
      // 16*1024 bf16 per half = 2048 x 16B
      for (int i = threadIdx.x; i < 2048; i += 256) {
        async_cp16(ldsBase + i * 16, (const char*)srcHi + i * 16);
        async_cp16(ldsBase + 32768 + i * 16, (const char*)srcLo + i * 16);
      }
      async_wait0();
    }
    __syncthreads();  // sA ready for all waves

    // ---- init accumulators with input-projection pre-activations
    const float* xpt = XP + (size_t)(t * B_SZ + bBase) * G4H;
    v8f acc[4];
#pragma unroll
    for (int gate = 0; gate < 4; ++gate)
#pragma unroll
      for (int r = 0; r < 8; ++r)
        acc[gate][r] = xpt[(size_t)(r + hl * 8) * G4H + gate * H_SZ + j];

    // ---- h(t-1) @ w_hh^T (bf16x2 split, 3 WMMA per term)
    for (int k0 = 0; k0 < H_SZ; k0 += 32) {
      v8bf a0 = *(const v8bf*)(sA + ln * H_SZ + k0 + hl * 16);
      v8bf a1 = *(const v8bf*)(sA + ln * H_SZ + k0 + 8 + hl * 16);
      v8bf a2 = *(const v8bf*)(sA + 16 * H_SZ + ln * H_SZ + k0 + hl * 16);
      v8bf a3 = *(const v8bf*)(sA + 16 * H_SZ + ln * H_SZ + k0 + 8 + hl * 16);
      v16bf Ah = __builtin_shufflevector(a0, a1, 0, 1, 2, 3, 4, 5, 6, 7, 8, 9,
                                         10, 11, 12, 13, 14, 15);
      v16bf Al = __builtin_shufflevector(a2, a3, 0, 1, 2, 3, 4, 5, 6, 7, 8, 9,
                                         10, 11, 12, 13, 14, 15);
#pragma unroll
      for (int gate = 0; gate < 4; ++gate) {
        const int g = gate * H_SZ + j;   // weight row / output column
        const v16bf Bh = *(const v16bf*)(Uhi + (size_t)g * H_SZ + k0 + hl * 16);
        const v16bf Bl = *(const v16bf*)(Ulo + (size_t)g * H_SZ + k0 + hl * 16);
        acc[gate] = wmma_bf16(Ah, Bh, acc[gate]);
        acc[gate] = wmma_bf16(Ah, Bl, acc[gate]);
        acc[gate] = wmma_bf16(Al, Bh, acc[gate]);
      }
    }

    // ---- cell update (register-local), emit h(t)
    bf16* dstHi = stage + (size_t)wp * STG_PAR;
    bf16* dstLo = dstHi + STG_HALF;
#pragma unroll
    for (int r = 0; r < 8; ++r) {
      float ig = sigmoidf_(acc[0][r]);
      float fg = sigmoidf_(acc[1][r]);
      float gg = tanhf(acc[2][r]);
      float og = sigmoidf_(acc[3][r]);
      float c = fg * cst[r] + ig * gg;
      cst[r] = c;
      float h = og * tanhf(c);

      const int bl = r + hl * 8;                 // local batch row 0..15
      const int srow = (bBase + bl) * H_SZ + j;  // staging index
      bf16 hh = (bf16)h;
      dstHi[srow] = hh;
      dstLo[srow] = (bf16)(h - (float)hh);

      const size_t orow = (size_t)(t * B_SZ + bBase + bl);
      if (!layer2) {
        Hout[orow * H_SZ + j] = h;
      } else {
        float v = h + Hres[orow * H_SZ + j];
        Hout[orow * H_SZ + j] = v;          // out11
        Out14[orow * 2048 + j] = v;         // out14 first half
      }
    }

    // ---- slab barrier: release our h(t), wait for the other 7 WGs
    __threadfence();       // make staging stores agent-visible
    __syncthreads();       // all threads of this WG have fenced
    if (threadIdx.x == 0)
      __hip_atomic_fetch_add(ctr, 1u, __ATOMIC_RELEASE,
                             __HIP_MEMORY_SCOPE_AGENT);
    const uint32_t target = 8u * (uint32_t)(t + 1);
    while (__hip_atomic_load(ctr, __ATOMIC_ACQUIRE,
                             __HIP_MEMORY_SCOPE_AGENT) < target)
      __builtin_amdgcn_s_sleep(1);
  }
}

// ---------------------------------------------------------------------------
extern "C" void kernel_launch(void* const* d_in, const int* in_sizes, int n_in,
                              void* d_out, int out_size, void* d_ws,
                              size_t ws_size, hipStream_t stream) {
  (void)in_sizes; (void)n_in; (void)out_size; (void)ws_size;

  const float* input4 = (const float*)d_in[0];
  const float* input0 = (const float*)d_in[1];
  const float* w_ih6  = (const float*)d_in[2];
  const float* w_hh6  = (const float*)d_in[3];
  const float* b_ih6  = (const float*)d_in[4];
  const float* b_hh6  = (const float*)d_in[5];
  const float* w_ih10 = (const float*)d_in[6];
  const float* w_hh10 = (const float*)d_in[7];
  const float* b_ih10 = (const float*)d_in[8];
  const float* b_hh10 = (const float*)d_in[9];

  float* out1  = (float*)d_out;                       // [ROWS, H]
  float* out11 = out1 + (size_t)ROWS * H_SZ;          // [ROWS, H]
  float* out14 = out11 + (size_t)ROWS * H_SZ;         // [ROWS, 2H]

  // Workspace carve-up (all 256B aligned)
  char* ws = (char*)d_ws;
  size_t off = 0;
  auto carve = [&](size_t bytes) -> void* {
    void* p = ws + off;
    off += (bytes + 255) & ~(size_t)255;
    return p;
  };
  const size_t nIh = (size_t)G4H * I_SZ;   // 8.39M
  const size_t nHh = (size_t)G4H * H_SZ;   // 4.19M
  bf16* w1ih_hi = (bf16*)carve(nIh * 2);
  bf16* w1ih_lo = (bf16*)carve(nIh * 2);
  bf16* w1hh_hi = (bf16*)carve(nHh * 2);
  bf16* w1hh_lo = (bf16*)carve(nHh * 2);
  bf16* w2ih_hi = (bf16*)carve(nIh * 2);
  bf16* w2ih_lo = (bf16*)carve(nIh * 2);
  bf16* w2hh_hi = (bf16*)carve(nHh * 2);
  bf16* w2hh_lo = (bf16*)carve(nHh * 2);
  float* xp = (float*)carve((size_t)ROWS * G4H * 4);  // 256 MB, reused per layer
  float* h1 = (float*)carve((size_t)ROWS * H_SZ * 4); // 64 MB (out7)
  bf16* stage1 = (bf16*)carve((size_t)STG_TOT * 2);   // 1 MB, layer-1 h pingpong
  bf16* stage2 = (bf16*)carve((size_t)STG_TOT * 2);   // 1 MB, layer-2 h pingpong
  uint32_t* syncc = (uint32_t*)carve(64);             // 8+8 slab counters

  // Zero staging parities + counters (stage1/stage2/sync are contiguous)
  zero32_kernel<<<1024, 256, 0, stream>>>((uint32_t*)stage1,
                                          (int)((2 * (size_t)STG_TOT * 2 + 64) / 4));

  // Static output pieces + weight splitting
  prep_out_kernel<<<2048, 256, 0, stream>>>(input4, out1, out14);
  cvt_split_kernel<<<4096, 256, 0, stream>>>(w_ih6, w1ih_hi, w1ih_lo, (int)nIh);
  cvt_split_kernel<<<4096, 256, 0, stream>>>(w_hh6, w1hh_hi, w1hh_lo, (int)nHh);
  cvt_split_kernel<<<4096, 256, 0, stream>>>(w_ih10, w2ih_hi, w2ih_lo, (int)nIh);
  cvt_split_kernel<<<4096, 256, 0, stream>>>(w_hh10, w2hh_hi, w2hh_lo, (int)nHh);

  dim3 ggrid(G4H / 256, ROWS / 64);  // 16 x 256

  // Layer 1: x = concat(input0, input4)
  gemm_xproj_kernel<<<ggrid, 256, 0, stream>>>(input0, input4, w1ih_hi,
                                               w1ih_lo, b_ih6, b_hh6, xp);
  lstm_rec_kernel<<<64, 256, 0, stream>>>(xp, w1hh_hi, w1hh_lo, nullptr, h1,
                                          nullptr, stage1, syncc, 0);

  // Layer 2: x = concat(out7, input4), out11 = lstm(...) + out7
  gemm_xproj_kernel<<<ggrid, 256, 0, stream>>>(h1, input4, w2ih_hi, w2ih_lo,
                                               b_ih10, b_hh10, xp);
  lstm_rec_kernel<<<64, 256, 0, stream>>>(xp, w2hh_hi, w2hh_lo, h1, out11,
                                          out14, stage2, syncc + 8, 1);
}
